// MCFNet_72052371358010
// MI455X (gfx1250) — compile-verified, matrix operand order checked by reading, not run on hardware
//
#include <hip/hip_runtime.h>
#include <stddef.h>

// ---------------------------------------------------------------------------
// CDNA5 / gfx1250 MCFNet forward. wave32, WMMA bf16 16x16x32, f32 accumulate.
// LDS tiles stored pre-swizzled in WMMA fragment order: each lane's 16 bf16
// operand elements are contiguous -> fragments load as 2x ds_load_b128.
// GEMM uses double-buffered GLOBAL_LOAD_ASYNC_TO_LDS_B128 (ASYNCcnt) staging.
// ---------------------------------------------------------------------------

typedef __attribute__((ext_vector_type(16))) __bf16        v16bf;
typedef __attribute__((ext_vector_type(8)))  float         v8f;
typedef __attribute__((ext_vector_type(8)))  unsigned int  v8u;

union FragCast { v8u u; v16bf b; };

__device__ __forceinline__ unsigned short f2bf(float f) {
  unsigned u = __float_as_uint(f);
  u += 0x7FFFu + ((u >> 16) & 1u);     // round-to-nearest-even
  return (unsigned short)(u >> 16);
}

__device__ __forceinline__ v8f vzero8() {
  v8f v = {0.f, 0.f, 0.f, 0.f, 0.f, 0.f, 0.f, 0.f};
  return v;
}

// Swizzle: element (k in [0,32), mn in [0,16)) of a 16x32 (A) / 32x16 (B)
// bf16 operand tile -> ushort index within a 512-element LDS tile, such that
// hardware lane L finds its 16 fragment elements contiguous at L*16.
__device__ __forceinline__ int swz(int k, int mn) {
  int lane = mn + (((k >> 3) & 1) << 4);
  int idx = ((k & 16) >> 1) + (k & 6) + (k & 1);
  return (lane << 4) + idx;
}

__device__ __forceinline__ v16bf load_frag(const unsigned short* p) {
  FragCast f;
  uint4 a = *(const uint4*)p;        // ds_load_b128
  uint4 b = *(const uint4*)(p + 8);  // ds_load_b128
  f.u[0] = a.x; f.u[1] = a.y; f.u[2] = a.z; f.u[3] = a.w;
  f.u[4] = b.x; f.u[5] = b.y; f.u[6] = b.z; f.u[7] = b.w;
  return f.b;
}

__device__ __forceinline__ v8f wmma_bf16(v16bf a, v16bf b, v8f c) {
  return __builtin_amdgcn_wmma_f32_16x16x32_bf16(
      false, a, false, b, (short)0, c, false, false);
}

// Async DMA: 16 bytes global -> LDS per lane, tracked by ASYNCcnt (GVS mode:
// SGPR64 base + per-lane VGPR byte offset; VDST VGPR = LDS byte address).
__device__ __forceinline__ void async_g2l_b128(unsigned lds_byte,
                                               const void* gbase,
                                               unsigned gbyte) {
  asm volatile("global_load_async_to_lds_b128 %0, %1, %2"
               :
               : "v"(lds_byte), "v"(gbyte),
                 "s"((unsigned long long)(size_t)gbase)
               : "memory");
}

// ---------------------------------------------------------------------------
// Generic GEMM: C[M,N] = act(A[M,K] @ W[K,N] + bias[N]).  ACT: 0=none, 1=SiLU
// Block tile 64x128, 256 threads = 8 waves, K step 32. M%64==0, N%128==0, K%32==0.
// Pipeline: async-DMA strip kt+1 into raw LDS buffer while computing kt.
// ---------------------------------------------------------------------------
template <int ACT>
__global__ __launch_bounds__(256) void gemm_bias_kernel(
    const float* __restrict__ A, const float* __restrict__ W,
    const float* __restrict__ bias, float* __restrict__ C,
    int M, int K, int N) {
  extern __shared__ char gsm[];
  float* araw = (float*)gsm;                         // 2 x 64x32 f32  (16 KB)
  float* wraw = (float*)(gsm + 16 * 1024);           // 2 x 32x128 f32 (32 KB)
  unsigned short* as_ = (unsigned short*)(gsm + 48 * 1024);  // 4 swizzled tiles
  unsigned short* bs_ = as_ + 4 * 512;                       // 8 swizzled tiles

  const int tid = threadIdx.x;
  const int wave = tid >> 5, lane = tid & 31;
  const int m0 = blockIdx.x * 64, n0 = blockIdx.y * 128;
  const int mt = wave >> 1;
  const int ntb = (wave & 1) * 4;
  const unsigned araw_lds = (unsigned)(size_t)araw;
  const unsigned wraw_lds = (unsigned)(size_t)wraw;

  v8f acc[4];
#pragma unroll
  for (int t = 0; t < 4; ++t) acc[t] = vzero8();

  // issue 6 async b128 copies (A: 2, W: 4) for strip kt into raw buffer `buf`
  auto issue_strip = [&](int kt, int buf) {
    const int kb = kt << 5;
    {
      unsigned lds = araw_lds + (unsigned)buf * 8192u +
                     ((unsigned)(tid >> 2) << 7) + ((unsigned)(tid & 3) << 5);
      unsigned gof = (unsigned)(((size_t)(m0 + (tid >> 2)) * K + kb) * 4) +
                     ((unsigned)(tid & 3) << 5);
      async_g2l_b128(lds, A, gof);
      async_g2l_b128(lds + 16, A, gof + 16);
    }
    {
      unsigned lds = wraw_lds + (unsigned)buf * 16384u +
                     ((unsigned)(tid >> 3) << 9) + ((unsigned)(tid & 7) << 6);
      unsigned gof = (unsigned)(((size_t)(kb + (tid >> 3)) * N + n0) * 4) +
                     ((unsigned)(tid & 7) << 6);
      async_g2l_b128(lds, W, gof);
      async_g2l_b128(lds + 16, W, gof + 16);
      async_g2l_b128(lds + 32, W, gof + 32);
      async_g2l_b128(lds + 48, W, gof + 48);
    }
  };

  const int nk = K >> 5;
  issue_strip(0, 0);
  for (int kt = 0; kt < nk; ++kt) {
    const int cur = kt & 1;
    if (kt + 1 < nk) {
      issue_strip(kt + 1, cur ^ 1);
      // 6 newer copies in flight; in-order completion => strip kt has landed
      asm volatile("s_wait_asynccnt 0x6" ::: "memory");
    } else {
      asm volatile("s_wait_asynccnt 0x0" ::: "memory");
    }
    __syncthreads();
    {  // convert raw f32 strips -> swizzled bf16 fragment tiles
      int r = tid >> 2, c = (tid & 3) << 3;
      const float* src = araw + cur * 2048 + r * 32 + c;
#pragma unroll
      for (int j = 0; j < 8; ++j)
        as_[((r >> 4) << 9) + swz(c + j, r & 15)] = f2bf(src[j]);
      int kk = tid >> 3, c2 = (tid & 7) << 4;
      const float* src2 = wraw + cur * 4096 + kk * 128 + c2;
#pragma unroll
      for (int j = 0; j < 16; ++j) {
        int n = c2 + j;
        bs_[((n >> 4) << 9) + swz(kk, n & 15)] = f2bf(src2[j]);
      }
    }
    __syncthreads();
    v16bf af = load_frag(&as_[(mt << 9) + (lane << 4)]);
#pragma unroll
    for (int t = 0; t < 4; ++t) {
      v16bf bf_ = load_frag(&bs_[((ntb + t) << 9) + (lane << 4)]);
      acc[t] = wmma_bf16(af, bf_, acc[t]);
    }
    __syncthreads();
  }
  // epilogue: C layout lane n = lane&15, row m = r + 8*(lane>>4)
#pragma unroll
  for (int t = 0; t < 4; ++t) {
    int col = n0 + (ntb + t) * 16 + (lane & 15);
    float bv = bias ? bias[col] : 0.f;
    int mbase = m0 + mt * 16 + ((lane >> 4) << 3);
#pragma unroll
    for (int r = 0; r < 8; ++r) {
      float x = acc[t][r] + bv;
      if (ACT == 1) x = x / (1.f + __expf(-x));  // SiLU
      C[(size_t)(mbase + r) * N + col] = x;
    }
  }
}

// ---------------------------------------------------------------------------
// Fused attention: O = softmax(Q K^T * 8) V  for one (b, h, 64-query block).
// Q,K,V,O: [B*S, 768] f32, head h occupies columns [h*HD, (h+1)*HD).
// Dynamic LDS: sc f32 [64][516] | psw bf16 (P, swizzled) | stage | qs.
// ---------------------------------------------------------------------------
template <int HD, int H>
__global__ __launch_bounds__(256) void attn_kernel(
    const float* __restrict__ Q, const float* __restrict__ K,
    const float* __restrict__ V, float* __restrict__ O) {
  constexpr int S = 512, D = 768;
  constexpr int SCW = S + 4;
  constexpr int STG = (S > HD ? S : HD) * 32;  // ushort elems, swizzled tiles
  extern __shared__ char smem[];
  float* sc = (float*)smem;                                   // 64*SCW f32
  unsigned short* psw = (unsigned short*)(smem + 64 * SCW * 4);  // 64 tiles
  unsigned short* stage = psw + 64 * 512;
  unsigned short* qs = stage + STG;                           // 4 tiles

  const int tid = threadIdx.x;
  const int wave = tid >> 5, lane = tid & 31;
  const int mt = wave >> 1, half = wave & 1;

  const int qb = blockIdx.x & 7;             // S/64 == 8
  const int h  = (blockIdx.x >> 3) % H;
  const int b  = blockIdx.x / (8 * H);
  const int coff = h * HD;
  const size_t qr0 = (size_t)(b * S + qb * 64) * D;
  const size_t kr0 = (size_t)(b * S) * D;

  // ---- Phase A: scores = Q K^T * 8 ----
  v8f acc[16];
#pragma unroll
  for (int j = 0; j < 16; ++j) acc[j] = vzero8();
  constexpr int NKT = HD / 32;
  for (int kt = 0; kt < NKT; ++kt) {
    const int kb = coff + kt * 32;
    {  // Q strip 64x32 -> swizzled qs
      int r = tid >> 2, c = (tid & 3) << 3;
      const float* src = Q + qr0 + (size_t)r * D + kb + c;
#pragma unroll
      for (int j = 0; j < 8; ++j)
        qs[((r >> 4) << 9) + swz(c + j, r & 15)] = f2bf(src[j]);
    }
#pragma unroll
    for (int kk = 0; kk < 2; ++kk) {  // K strip 512x32 -> 32 swizzled tiles
      int key = tid + (kk << 8);
      const float* src = K + kr0 + (size_t)key * D + kb;
#pragma unroll
      for (int d = 0; d < 32; ++d)
        stage[((key >> 4) << 9) + swz(d, key & 15)] = f2bf(src[d]);
    }
    __syncthreads();
    v16bf af = load_frag(&qs[(mt << 9) + (lane << 4)]);
#pragma unroll
    for (int j = 0; j < 16; ++j) {
      int nt = (half << 4) + j;
      v16bf bf_ = load_frag(&stage[(nt << 9) + (lane << 4)]);
      acc[j] = wmma_bf16(af, bf_, acc[j]);
    }
    __syncthreads();
  }
  {  // spill scores (x8, per reference) to LDS f32
    int mrow = mt * 16 + ((lane >> 4) << 3);
#pragma unroll
    for (int j = 0; j < 16; ++j) {
      int colb = (half << 8) + j * 16 + (lane & 15);
#pragma unroll
      for (int r = 0; r < 8; ++r)
        sc[(mrow + r) * SCW + colb] = acc[j][r] * 8.0f;
    }
  }
  __syncthreads();

  // ---- row softmax over 512 keys (4 lanes per row); emit P as bf16 swizzled
  {
    int row = tid >> 2, seg = tid & 3;
    float* rp = sc + row * SCW + seg * 128;
    float mx = -3.4e38f;
    for (int c = 0; c < 128; ++c) mx = fmaxf(mx, rp[c]);
    mx = fmaxf(mx, __shfl_xor(mx, 1));
    mx = fmaxf(mx, __shfl_xor(mx, 2));
    float sum = 0.f;
    for (int c = 0; c < 128; ++c) { float e = __expf(rp[c] - mx); rp[c] = e; sum += e; }
    sum += __shfl_xor(sum, 1);
    sum += __shfl_xor(sum, 2);
    float inv = 1.f / sum;
    int tmrow = (row >> 4) << 4;
    for (int c = 0; c < 128; ++c) {
      int col = seg * 128 + c;
      int tile = tmrow + (col >> 5);
      psw[(tile << 9) + swz(col & 31, row & 15)] = f2bf(rp[c] * inv);
    }
  }
  __syncthreads();

  // ---- Phase C: O = P @ V ----
  constexpr int MYNT = HD / 32;  // n-tiles per wave (interleaved by `half`)
  v8f oacc[MYNT];
#pragma unroll
  for (int j = 0; j < MYNT; ++j) oacc[j] = vzero8();
  for (int kt2 = 0; kt2 < S / 32; ++kt2) {
    {  // V strip 32 keys x HD dims -> HD/16 swizzled tiles (k=key, n=dim)
      int k = tid >> 3;
      int d0 = (tid & 7) * (HD / 8);
      const float* src = V + kr0 + (size_t)(kt2 * 32 + k) * D + coff + d0;
#pragma unroll
      for (int d = 0; d < HD / 8; ++d) {
        int dd = d0 + d;
        stage[((dd >> 4) << 9) + swz(k, dd & 15)] = f2bf(src[d]);
      }
    }
    __syncthreads();
    v16bf af = load_frag(&psw[((mt * 16 + kt2) << 9) + (lane << 4)]);
#pragma unroll
    for (int j = 0; j < MYNT; ++j) {
      int nt = half + 2 * j;
      v16bf bf_ = load_frag(&stage[(nt << 9) + (lane << 4)]);
      oacc[j] = wmma_bf16(af, bf_, oacc[j]);
    }
    __syncthreads();
  }
  {  // write O
    int mbase = qb * 64 + mt * 16 + ((lane >> 4) << 3);
#pragma unroll
    for (int j = 0; j < MYNT; ++j) {
      int col = coff + (half + 2 * j) * 16 + (lane & 15);
#pragma unroll
      for (int r = 0; r < 8; ++r)
        O[(size_t)(b * S + mbase + r) * D + col] = oacc[j][r];
    }
  }
}

// ---------------------------------------------------------------------------
// Elementwise helpers
// ---------------------------------------------------------------------------
__global__ __launch_bounds__(256) void gather_kernel(
    const float* __restrict__ table, const int* __restrict__ ids,
    float* __restrict__ out) {
  const int D = 768;
  size_t e = (size_t)blockIdx.x * 256 + threadIdx.x;
  size_t row = e / D;
  int c = (int)(e % D);
  out[e] = table[(size_t)ids[row] * D + c];
}

__global__ __launch_bounds__(256) void rope_kernel(float* __restrict__ X) {
  const int D = 768, S = 512, HALF = 384;
  size_t e = (size_t)blockIdx.x * 256 + threadIdx.x;  // over 16384*384
  size_t row = e / HALF;
  int i = (int)(e % HALF);
  int pos = (int)(row % S);
  float theta = __expf((float)i * (-2.0f * 9.210340371976184f / 768.0f));
  float ang = (float)pos * theta;
  float s, c;
  __sincosf(ang, &s, &c);
  size_t o = row * D + 2 * i;
  float x0 = X[o], x1 = X[o + 1];
  X[o]     = x0 * c - x1 * s;
  X[o + 1] = x1 * c + x0 * s;
}

__global__ __launch_bounds__(256) void concat_kernel(
    const float* __restrict__ X, const float* __restrict__ Y,
    float* __restrict__ out) {
  size_t e = (size_t)blockIdx.x * 256 + threadIdx.x;  // over 16384*1536
  size_t row = e / 1536;
  int c = (int)(e % 1536);
  out[e] = (c < 768) ? X[row * 768 + c] : Y[row * 768 + (c - 768)];
}

// gate(x) = x * softmax(x, -1) + x ; one block per row of 768
__global__ __launch_bounds__(256) void gate_kernel(
    const float* __restrict__ in, float* __restrict__ out) {
  __shared__ float red[8];
  const int D = 768;
  size_t base = (size_t)blockIdx.x * D;
  int tid = threadIdx.x;
  float v[3];
  float mx = -3.4e38f;
#pragma unroll
  for (int i = 0; i < 3; ++i) { v[i] = in[base + tid + 256 * i]; mx = fmaxf(mx, v[i]); }
#pragma unroll
  for (int o = 16; o >= 1; o >>= 1) mx = fmaxf(mx, __shfl_xor(mx, o));
  if ((tid & 31) == 0) red[tid >> 5] = mx;
  __syncthreads();
  mx = red[0];
#pragma unroll
  for (int i = 1; i < 8; ++i) mx = fmaxf(mx, red[i]);
  __syncthreads();
  float e[3];
  float s = 0.f;
#pragma unroll
  for (int i = 0; i < 3; ++i) { e[i] = __expf(v[i] - mx); s += e[i]; }
#pragma unroll
  for (int o = 16; o >= 1; o >>= 1) s += __shfl_xor(s, o);
  if ((tid & 31) == 0) red[tid >> 5] = s;
  __syncthreads();
  s = 0.f;
#pragma unroll
  for (int i = 0; i < 8; ++i) s += red[i];
  float inv = 1.f / s;
#pragma unroll
  for (int i = 0; i < 3; ++i) out[base + tid + 256 * i] = v[i] * (e[i] * inv) + v[i];
}

__global__ __launch_bounds__(256) void sum4_kernel(
    const float* __restrict__ a, const float* __restrict__ b,
    const float* __restrict__ c, const float* __restrict__ d,
    float* __restrict__ o) {
  size_t e = (size_t)blockIdx.x * 256 + threadIdx.x;
  o[e] = a[e] + b[e] + c[e] + d[e];
}

// ---------------------------------------------------------------------------
// Orchestration
// ---------------------------------------------------------------------------
extern "C" void kernel_launch(void* const* d_in, const int* in_sizes, int n_in,
                              void* d_out, int out_size, void* d_ws, size_t ws_size,
                              hipStream_t stream) {
  (void)in_sizes; (void)n_in; (void)out_size; (void)ws_size;
  const float* text    = (const float*)d_in[0];
  const int*   vis_ids = (const int*)d_in[1];
  const int*   ac_ids  = (const int*)d_in[2];
  const float* vis_tab = (const float*)d_in[3];
  const float* ac_tab  = (const float*)d_in[4];
  const float *hv_wq = (const float*)d_in[5],  *hv_bq = (const float*)d_in[6];
  const float *hv_wk = (const float*)d_in[7],  *hv_bk = (const float*)d_in[8];
  const float *hv_wv = (const float*)d_in[9],  *hv_bv = (const float*)d_in[10];
  const float *ha_wq = (const float*)d_in[11], *ha_bq = (const float*)d_in[12];
  const float *ha_wk = (const float*)d_in[13], *ha_bk = (const float*)d_in[14];
  const float *ha_wv = (const float*)d_in[15], *ha_bv = (const float*)d_in[16];
  const float *ht_wq = (const float*)d_in[17], *ht_bq = (const float*)d_in[18];
  const float *ht_wk = (const float*)d_in[19], *ht_bk = (const float*)d_in[20];
  const float *ht_wv = (const float*)d_in[21], *ht_bv = (const float*)d_in[22];
  const float *ht_fcw = (const float*)d_in[23], *ht_fcb = (const float*)d_in[24];
  const float *ffn_w1 = (const float*)d_in[25], *ffn_b1 = (const float*)d_in[26];
  const float *ffn_w2 = (const float*)d_in[27], *ffn_b2 = (const float*)d_in[28];
  const float *cat_w  = (const float*)d_in[29], *cat_b  = (const float*)d_in[30];

  const int BS = 16384;                 // 32*512 rows
  const size_t MD = (size_t)BS * 768;
  float* A = (float*)d_ws;
  float* Bb = A + MD;
  float* Cc = Bb + MD;
  float* Dd = Cc + MD;
  float* E = Dd + MD;
  float* F = E + MD;
  float* G = F + MD;
  float* CAT = G + MD;                  // BS x 1536

  const int EW = (int)(MD / 256);       // 49152
  const int EW2 = (int)(MD * 2 / 256);  // 98304
  const int EWR = (int)(MD / 2 / 256);  // 24576 (rope: one thread per pair)

  // dynamic LDS sizes
  const size_t SHMG = 48 * 1024 + 12 * 1024;  // gemm: raw dbl-buf + swizzled
  const size_t SHM1 = 64 * 516 * 4 + 64 * 512 * 2 + (size_t)768 * 32 * 2 + 4 * 512 * 2;  // 250880
  const size_t SHM8 = 64 * 516 * 4 + 64 * 512 * 2 + (size_t)512 * 32 * 2 + 4 * 512 * 2;  // 234496
  (void)hipFuncSetAttribute((const void*)gemm_bias_kernel<0>,
                            hipFuncAttributeMaxDynamicSharedMemorySize, (int)SHMG);
  (void)hipFuncSetAttribute((const void*)gemm_bias_kernel<1>,
                            hipFuncAttributeMaxDynamicSharedMemorySize, (int)SHMG);
  (void)hipFuncSetAttribute((const void*)attn_kernel<768, 1>,
                            hipFuncAttributeMaxDynamicSharedMemorySize, (int)SHM1);
  (void)hipFuncSetAttribute((const void*)attn_kernel<96, 8>,
                            hipFuncAttributeMaxDynamicSharedMemorySize, (int)SHM8);

  dim3 gg(BS / 64, 768 / 128);
  auto GEMM = [&](const float* Ain, const float* Win, const float* bin,
                  float* Cout, int Kdim, int act) {
    if (act) gemm_bias_kernel<1><<<gg, 256, SHMG, stream>>>(Ain, Win, bin, Cout, BS, Kdim, 768);
    else     gemm_bias_kernel<0><<<gg, 256, SHMG, stream>>>(Ain, Win, bin, Cout, BS, Kdim, 768);
  };
  auto ROPE = [&](float* X) { rope_kernel<<<EWR, 256, 0, stream>>>(X); };
  auto ATTN1 = [&](const float* q, const float* k, const float* v, float* o) {
    attn_kernel<768, 1><<<32 * 1 * 8, 256, SHM1, stream>>>(q, k, v, o);
  };
  auto FFN = [&](const float* x, float* tmp, float* y) {  // y = silu(x@w1+b1)@w2+b2
    GEMM(x, ffn_w1, ffn_b1, tmp, 768, 1);
    GEMM(tmp, ffn_w2, ffn_b2, y, 768, 0);
  };

  // embeddings: A = visual embed, B = acoustic embed
  gather_kernel<<<EW, 256, 0, stream>>>(vis_tab, vis_ids, A);
  gather_kernel<<<EW, 256, 0, stream>>>(ac_tab, ac_ids, Bb);

  // visual_ = hv(text, A) -> F
  GEMM(text, hv_wq, hv_bq, Cc, 768, 0); ROPE(Cc);
  GEMM(A,    hv_wk, hv_bk, Dd, 768, 0); ROPE(Dd);
  GEMM(A,    hv_wv, hv_bv, E,  768, 0);
  ATTN1(Cc, Dd, E, F);

  // acoustic_ = ha(text, B) -> A
  GEMM(text, ha_wq, ha_bq, Cc, 768, 0); ROPE(Cc);
  GEMM(Bb,   ha_wk, ha_bk, Dd, 768, 0); ROPE(Dd);
  GEMM(Bb,   ha_wv, ha_bv, E,  768, 0);
  ATTN1(Cc, Dd, E, A);

  // F1 = ffn(gate(concat(visual_, acoustic_) @ cat_w + cat_b)) -> B
  concat_kernel<<<EW2, 256, 0, stream>>>(F, A, CAT);
  GEMM(CAT, cat_w, cat_b, Bb, 1536, 0);
  gate_kernel<<<BS, 256, 0, stream>>>(Bb, G);
  FFN(G, Cc, Bb);                       // B = F1 (kept for F3 and final sum)

  // v = hv(visual_=F, acoustic_=A) -> G
  GEMM(F, hv_wq, hv_bq, Cc, 768, 0); ROPE(Cc);
  GEMM(A, hv_wk, hv_bk, Dd, 768, 0); ROPE(Dd);
  GEMM(A, hv_wv, hv_bv, E,  768, 0);
  ATTN1(Cc, Dd, E, G);

  // a = ha(acoustic_=A, visual_=F) -> A
  GEMM(A, ha_wq, ha_bq, Cc, 768, 0); ROPE(Cc);
  GEMM(F, ha_wk, ha_bk, Dd, 768, 0); ROPE(Dd);
  GEMM(F, ha_wv, ha_bv, E,  768, 0);
  ATTN1(Cc, Dd, E, A);

  // F2 = ffn(gate(concat(v, a) @ cat_w + cat_b)) -> C
  concat_kernel<<<EW2, 256, 0, stream>>>(G, A, CAT);
  GEMM(CAT, cat_w, cat_b, Cc, 1536, 0);
  gate_kernel<<<BS, 256, 0, stream>>>(Cc, Dd);
  GEMM(Dd, ffn_w1, ffn_b1, E, 768, 1);
  GEMM(E, ffn_w2, ffn_b2, Cc, 768, 0);  // C = F2

  // F3 = ffn(gate(attn8(F1, text) @ ht_fcw + ht_fcb)) -> A
  GEMM(Bb,   ht_wq, ht_bq, Dd, 768, 0); ROPE(Dd);
  GEMM(text, ht_wk, ht_bk, E,  768, 0); ROPE(E);
  GEMM(text, ht_wv, ht_bv, F,  768, 0);
  attn_kernel<96, 8><<<32 * 8 * 8, 256, SHM8, stream>>>(Dd, E, F, G);
  GEMM(G, ht_fcw, ht_fcb, A, 768, 0);
  gate_kernel<<<BS, 256, 0, stream>>>(A, Dd);
  GEMM(Dd, ffn_w1, ffn_b1, E, 768, 1);
  GEMM(E, ffn_w2, ffn_b2, A, 768, 0);   // A = F3

  // out = gate(text + F1 + F2 + F3)
  sum4_kernel<<<EW, 256, 0, stream>>>(text, Bb, Cc, A, Dd);
  gate_kernel<<<BS, 256, 0, stream>>>(Dd, (float*)d_out);
}
